// MultiScaleDefAttn3D_85289460564233
// MI455X (gfx1250) — compile-verified, matrix operand order checked by reading, not run on hardware
//
#include <hip/hip_runtime.h>

// ---------------------------------------------------------------------------
// Problem constants (fixed by the reference)
// ---------------------------------------------------------------------------
#define BB 1
#define VV 6
#define DNUM 4
#define QSUB 1024
#define NN (DNUM * QSUB)          // 4096
#define EE 256
#define HEADS 8
#define LVLS 4
#define PTS 4
#define EH (EE / HEADS)           // 32
#define BVN (VV * NN)             // 24576 query rows
#define OFFC (HEADS * LVLS * PTS * 3)  // 384
#define WGTC (HEADS * LVLS * PTS)      // 128

__device__ __constant__ int   c_LD[4]   = {4, 4, 4, 4};
__device__ __constant__ int   c_LH[4]   = {64, 32, 16, 8};
__device__ __constant__ int   c_LW[4]   = {176, 88, 44, 22};
__device__ __constant__ int   c_LOFF[4] = {0, 45056, 56320, 59136};

typedef __attribute__((ext_vector_type(16))) __bf16 v16bf;
typedef __attribute__((ext_vector_type(8)))  float  v8f;

union V16U {
    uint4 u[2];
    v16bf v;
};

static __device__ __forceinline__ unsigned short f2bf(float f) {
    unsigned int u = __float_as_uint(f);
    unsigned int r = u + 0x7FFFu + ((u >> 16) & 1u);   // round-to-nearest-even
    return (unsigned short)(r >> 16);
}

// ---------------------------------------------------------------------------
// Kernel 0: convert W_offset/W_weight/W_proj (f32, [K][Ncol]) into bf16
// column-major tables WT[Ncol][K] so WMMA B-operand loads are contiguous.
// ---------------------------------------------------------------------------
__global__ __launch_bounds__(256) void k_prep(
    const float* __restrict__ W_offset, const float* __restrict__ W_weight,
    const float* __restrict__ W_proj,
    unsigned short* __restrict__ WT_off, unsigned short* __restrict__ WT_wgt,
    unsigned short* __restrict__ WT_proj)
{
    int i = blockIdx.x * 256 + threadIdx.x;
    if (i < OFFC * EE) {                       // 98304
        int n = i >> 8, k = i & 255;
        WT_off[i] = f2bf(W_offset[k * OFFC + n]);
    } else if (i < OFFC * EE + WGTC * EE) {    // +32768
        int j = i - OFFC * EE;
        int n = j >> 8, k = j & 255;
        WT_wgt[j] = f2bf(W_weight[k * WGTC + n]);
    } else if (i < OFFC * EE + WGTC * EE + EE * EE) {  // +65536
        int j = i - OFFC * EE - WGTC * EE;
        int n = j >> 8, k = j & 255;
        WT_proj[j] = f2bf(W_proj[k * EE + n]);
    }
}

// ---------------------------------------------------------------------------
// Kernel 1: fused per-16-query tile:
//   WMMA projections (offset 384, attn logits 128), softmax-16 per head,
//   depth dot-product, then deformable trilinear sampling + attn-weighted
//   accumulation. Output: SAMPOUT[gq][e], DEP[gq].
// ---------------------------------------------------------------------------
__global__ __launch_bounds__(256) void k_fused(
    const float* __restrict__ query, const float* __restrict__ value,
    const float* __restrict__ point, const unsigned char* __restrict__ valid,
    const float* __restrict__ b_offset, const float* __restrict__ b_weight,
    const float* __restrict__ W_depth, const float* __restrict__ b_depth,
    const unsigned short* __restrict__ WT_off,
    const unsigned short* __restrict__ WT_wgt,
    float* __restrict__ SAMPOUT, float* __restrict__ DEP)
{
    __shared__ __align__(16) unsigned short Qs[16 * EE];   // 8 KB bf16 query tile
    __shared__ float OFFs[16 * OFFC];                      // 24 KB
    __shared__ float WLs[16 * WGTC];                       // 8 KB

    const int tid   = threadIdx.x;
    const long qbase = (long)blockIdx.x * 16;

    // Stage query tile as bf16
    for (int i = tid; i < 16 * EE; i += 256)
        Qs[i] = f2bf(query[qbase * EE + i]);
    __syncthreads();

    // ---- WMMA projections: 32 column tiles over 8 waves (4 each) ----
    const int wave = tid >> 5, lane = tid & 31;
    const int g = lane >> 4, nl = lane & 15;   // lane group / row-col in tile

    for (int t = wave * 4; t < wave * 4 + 4; ++t) {
        const unsigned short* WT;
        const float* bias;
        float* dst;
        int dstride, colbase;
        if (t < 24) { WT = WT_off; bias = b_offset; dst = OFFs; dstride = OFFC; colbase = t * 16; }
        else        { WT = WT_wgt; bias = b_weight; dst = WLs;  dstride = WGTC; colbase = (t - 24) * 16; }

        v8f c = {0.f, 0.f, 0.f, 0.f, 0.f, 0.f, 0.f, 0.f};
        const unsigned short* wcol = WT + (colbase + nl) * EE;
        #pragma unroll
        for (int k0 = 0; k0 < EE; k0 += 32) {
            V16U a, b;
            a.u[0] = *(const uint4*)&Qs[nl * EE + k0 + g * 8];
            a.u[1] = *(const uint4*)&Qs[nl * EE + k0 + 16 + g * 8];
            b.u[0] = *(const uint4*)(wcol + k0 + g * 8);
            b.u[1] = *(const uint4*)(wcol + k0 + 16 + g * 8);
            c = __builtin_amdgcn_wmma_f32_16x16x32_bf16(
                    false, a.v, false, b.v, (short)0, c, false, false);
        }
        const int col = colbase + nl;
        const float bv = bias[col];
        const int m0 = g * 8;
        #pragma unroll
        for (int r = 0; r < 8; ++r)
            dst[(m0 + r) * dstride + col] = c[r] + bv;
    }

    // ---- depth: one dot product per query row ----
    if (tid < 16) {
        const float* qrow = query + (qbase + tid) * EE;
        float acc = 0.f;
        for (int k = 0; k < EE; ++k) acc += qrow[k] * W_depth[k];
        DEP[qbase + tid] = acc + b_depth[0];
    }
    __syncthreads();

    // ---- softmax over 16 (lvl,pt) logits per (query, head) ----
    if (tid < 128) {
        int q = tid >> 3, h = tid & 7;
        float* wl = &WLs[q * WGTC + h * 16];
        float mx = wl[0];
        #pragma unroll
        for (int i = 1; i < 16; ++i) mx = fmaxf(mx, wl[i]);
        float e[16], s = 0.f;
        #pragma unroll
        for (int i = 0; i < 16; ++i) { e[i] = __expf(wl[i] - mx); s += e[i]; }
        float inv = 1.f / s;
        #pragma unroll
        for (int i = 0; i < 16; ++i) wl[i] = e[i] * inv;
    }
    __syncthreads();

    // ---- deformable sampling: thread owns (q, head, 16-channel half) ----
    const int q = tid >> 4, sub = tid & 15;
    const int h = sub >> 1, half = sub & 1;
    const long gq = qbase + q;
    const int v = (int)(gq >> 12);          // / NN
    const int n = (int)(gq & (NN - 1));
    const float px = point[gq * 3 + 0];
    const float py = point[gq * 3 + 1];
    const float pz = point[gq * 3 + 2];

    float acc[16];
    #pragma unroll
    for (int c = 0; c < 16; ++c) acc[c] = 0.f;
    const float* vch = value + h * EH + half * 16;

    #pragma unroll
    for (int lvl = 0; lvl < LVLS; ++lvl) {
        const int Di = c_LD[lvl], Hi = c_LH[lvl], Wi = c_LW[lvl];
        const float Df = (float)Di, Hf = (float)Hi, Wf = (float)Wi;
        const int loff = c_LOFF[lvl];
        for (int p = 0; p < PTS; ++p) {
            const int oi = ((h * LVLS + lvl) * PTS + p) * 3;
            const float lx = px + OFFs[q * OFFC + oi + 0] / Wf;
            const float ly = py + OFFs[q * OFFC + oi + 1] / Hf;
            const float lz = pz + OFFs[q * OFFC + oi + 2] / Df;
            // grid = loc*2-1 ; ix = ((gx+1)*W-1)*0.5 = loc_x*W - 0.5
            const float ix = lx * Wf - 0.5f;
            const float iy = ly * Hf - 0.5f;
            const float iz = lz * Df - 0.5f;
            const float xf = floorf(ix), yf = floorf(iy), zf = floorf(iz);
            const float fx = ix - xf, fy = iy - yf, fz = iz - zf;
            const int x0 = (int)xf, y0 = (int)yf, z0 = (int)zf;
            const float aw = WLs[q * WGTC + h * 16 + lvl * PTS + p];

            #pragma unroll
            for (int dz = 0; dz < 2; ++dz) {
                const int zc = z0 + dz;
                if ((unsigned)zc >= (unsigned)Di) continue;
                const float wz = dz ? fz : 1.f - fz;
                #pragma unroll
                for (int dy = 0; dy < 2; ++dy) {
                    const int yc = y0 + dy;
                    if ((unsigned)yc >= (unsigned)Hi) continue;
                    const float wy = dy ? fy : 1.f - fy;
                    #pragma unroll
                    for (int dx = 0; dx < 2; ++dx) {
                        const int xc = x0 + dx;
                        if ((unsigned)xc >= (unsigned)Wi) continue;
                        const float w = aw * wz * wy * (dx ? fx : 1.f - fx);
                        const long row = loff + (long)(zc * Hi + yc) * Wi + xc;
                        const float* src = vch + row * EE;
                        #pragma unroll
                        for (int c = 0; c < 16; ++c)
                            acc[c] = fmaf(w, src[c], acc[c]);
                    }
                }
            }
        }
    }

    const float mval = valid[v * NN + n] ? 1.f : 0.f;
    float* op = SAMPOUT + gq * EE + h * EH + half * 16;
    #pragma unroll
    for (int c = 0; c < 16; ++c) op[c] = acc[c] * mval;
}

// ---------------------------------------------------------------------------
// Kernel 2: depth-bin softmax aggregation over DNUM=4 groups.
// ---------------------------------------------------------------------------
__global__ __launch_bounds__(256) void k_depthagg(
    const float* __restrict__ SAMPOUT, const float* __restrict__ DEP,
    const unsigned char* __restrict__ valid, float* __restrict__ AGG)
{
    const int bid = blockIdx.x;
    const int v = bid >> 10, qs = bid & (QSUB - 1);
    const int e = threadIdx.x;

    float l[4];
    #pragma unroll
    for (int d = 0; d < DNUM; ++d) {
        const int n = d * QSUB + qs;
        const float dep = DEP[v * NN + n];
        l[d] = valid[v * NN + n] ? dep : -3.4028234663852886e38f;
    }
    float mx = fmaxf(fmaxf(l[0], l[1]), fmaxf(l[2], l[3]));
    float w[4], s = 0.f;
    #pragma unroll
    for (int d = 0; d < DNUM; ++d) { w[d] = __expf(l[d] - mx); s += w[d]; }
    const float inv = 1.f / s;

    float acc = 0.f;
    #pragma unroll
    for (int d = 0; d < DNUM; ++d) {
        const long gq = (long)v * NN + d * QSUB + qs;
        acc = fmaf(w[d] * inv, SAMPOUT[gq * EE + e], acc);
    }
    AGG[((long)v * QSUB + qs) * EE + e] = acc;
}

// ---------------------------------------------------------------------------
// Kernel 3: projection GEMM [6144,256] x [256,256] via bf16 WMMA + bias,
// broadcast over DNUM into d_out[B,V,DNUM,QSUB,E].
// ---------------------------------------------------------------------------
__global__ __launch_bounds__(256) void k_proj(
    const float* __restrict__ AGG, const unsigned short* __restrict__ WT_proj,
    const float* __restrict__ b_proj, float* __restrict__ out)
{
    __shared__ __align__(16) unsigned short As[16 * EE];

    const int tid = threadIdx.x;
    const long rbase = (long)blockIdx.x * 16;

    for (int i = tid; i < 16 * EE; i += 256)
        As[i] = f2bf(AGG[rbase * EE + i]);
    __syncthreads();

    const int wave = tid >> 5, lane = tid & 31;
    const int g = lane >> 4, nl = lane & 15;

    for (int ct = wave * 2; ct < wave * 2 + 2; ++ct) {
        v8f c = {0.f, 0.f, 0.f, 0.f, 0.f, 0.f, 0.f, 0.f};
        const unsigned short* wcol = WT_proj + (ct * 16 + nl) * EE;
        #pragma unroll
        for (int k0 = 0; k0 < EE; k0 += 32) {
            V16U a, b;
            a.u[0] = *(const uint4*)&As[nl * EE + k0 + g * 8];
            a.u[1] = *(const uint4*)&As[nl * EE + k0 + 16 + g * 8];
            b.u[0] = *(const uint4*)(wcol + k0 + g * 8);
            b.u[1] = *(const uint4*)(wcol + k0 + 16 + g * 8);
            c = __builtin_amdgcn_wmma_f32_16x16x32_bf16(
                    false, a.v, false, b.v, (short)0, c, false, false);
        }
        const int col = ct * 16 + nl;
        const float bv = b_proj[col];
        const int m0 = g * 8;
        #pragma unroll
        for (int r = 0; r < 8; ++r) {
            const long rm = rbase + m0 + r;      // row in [6144]
            const int v = (int)(rm >> 10), qs = (int)(rm & (QSUB - 1));
            const float val = c[r] + bv;
            #pragma unroll
            for (int d = 0; d < DNUM; ++d)
                out[(((long)(v * DNUM + d) * QSUB + qs) * EE) + col] = val;
        }
    }
}

// ---------------------------------------------------------------------------
// Launch
// ---------------------------------------------------------------------------
extern "C" void kernel_launch(void* const* d_in, const int* in_sizes, int n_in,
                              void* d_out, int out_size, void* d_ws, size_t ws_size,
                              hipStream_t stream)
{
    const float* query    = (const float*)d_in[0];
    const float* value    = (const float*)d_in[1];
    const float* point    = (const float*)d_in[2];
    const unsigned char* valid = (const unsigned char*)d_in[3];
    // d_in[4] = shape (compile-time constant here)
    const float* W_offset = (const float*)d_in[5];
    const float* b_offset = (const float*)d_in[6];
    const float* W_weight = (const float*)d_in[7];
    const float* b_weight = (const float*)d_in[8];
    const float* W_depth  = (const float*)d_in[9];
    const float* b_depth  = (const float*)d_in[10];
    const float* W_proj   = (const float*)d_in[11];
    const float* b_proj   = (const float*)d_in[12];

    char* ws = (char*)d_ws;
    float* SAMPOUT = (float*)(ws + 0);                       // 24576*256*4 = 25165824
    float* AGG     = (float*)(ws + 25165824);                // 6144*256*4  = 6291456
    float* DEP     = (float*)(ws + 31457280);                // 24576*4     = 98304
    unsigned short* WT_off  = (unsigned short*)(ws + 31555584); // 384*256*2 = 196608
    unsigned short* WT_wgt  = (unsigned short*)(ws + 31752192); // 128*256*2 = 65536
    unsigned short* WT_proj = (unsigned short*)(ws + 31817728); // 256*256*2 = 131072
    float* out = (float*)d_out;

    k_prep<<<768, 256, 0, stream>>>(W_offset, W_weight, W_proj,
                                    WT_off, WT_wgt, WT_proj);
    k_fused<<<BVN / 16, 256, 0, stream>>>(query, value, point, valid,
                                          b_offset, b_weight, W_depth, b_depth,
                                          WT_off, WT_wgt, SAMPOUT, DEP);
    k_depthagg<<<VV * QSUB, 256, 0, stream>>>(SAMPOUT, DEP, valid, AGG);
    k_proj<<<(VV * QSUB) / 16, 256, 0, stream>>>(AGG, WT_proj, b_proj, out);
}
